// SentenceEncoder_24335284699633
// MI455X (gfx1250) — compile-verified, matrix operand order checked by reading, not run on hardware
//
#include <hip/hip_runtime.h>
#include <hip/hip_bf16.h>

// ---------------------------------------------------------------------------
// SentenceEncoder on MI455X (gfx1250, wave32, WMMA)
//
// Pipeline (bf16 operands / f32 accumulate via v_wmma_f32_16x16x32_bf16):
//   1. convert emb -> bf16 padded K=320 ; W_ih -> bf16 padded ; W_hh -> bf16
//   2. x_proj[65536, 2048] = gather(emb)·W_ih^T + b_ih + b_hh   (WMMA GEMM)
//   3. 128 sequential LSTM steps (question+sentence fused, 512 rows):
//        gates = h·W_hh^T + x_proj[l]  (WMMA)  ;  cell update (VALU)
//   4. attn = _i·_t^T per batch (WMMA), row/col softmax,
//      fused (softmax·V + max-reduce) kernels (WMMA + shfl/LDS max)
//   5. tiny classifier + sigmoid
//
// All WMMA K-loops are software-pipelined with double-buffered fragment
// registers so loads for chunk k+1 overlap the matrix ops of chunk k
// (avoids s_wait_loadcnt 0 stalls in front of every v_wmma).
// Weights (W_ih 1.3MB, W_hh 2MB bf16) live in the 192MB L2, so direct
// global fragment loads are the right data path; no LDS staging needed.
// ---------------------------------------------------------------------------

typedef __bf16 bf16;
typedef __attribute__((ext_vector_type(16))) __bf16 v16bf;
typedef __attribute__((ext_vector_type(8)))  __bf16 v8bf;
typedef __attribute__((ext_vector_type(8)))  float  v8f;

constexpr int kVocab = 91557;
constexpr int kWD    = 300;   // word dim
constexpr int kWDP   = 320;   // padded to multiple of 32 for WMMA K loop
constexpr int kH     = 512;   // hidden
constexpr int kG     = 2048;  // 4*hidden (gates)
constexpr int kB     = 256;   // batch
constexpr int kSeq   = 128;   // sequence length
constexpr int kR     = 512;   // fused rows per LSTM step (2 sequences * batch)
constexpr int kMXP   = kSeq * kR; // 65536 x_proj rows

// ---------------------------------------------------------------------------
// WMMA helpers
// ---------------------------------------------------------------------------
__device__ __forceinline__ v8f wmma_bf16(v16bf a, v16bf b, v8f c) {
  // D = A(16x32 bf16) * B(32x16 bf16) + C(16x16 f32)
  return __builtin_amdgcn_wmma_f32_16x16x32_bf16(
      /*neg_a=*/false, a, /*neg_b=*/false, b,
      /*c_mod=*/(short)0, c, /*reuse_a=*/false, /*reuse_b=*/false);
}

// A-fragment (and symmetric B^T-fragment) loader from a per-lane row pointer.
// ISA layout: lane = row (lane&15), khalf = lane>>4; VGPRs 0..3 hold
// K = khalf*8 .. +7, VGPRs 4..7 hold K = 16 + khalf*8 .. +7 -> two 16B loads.
__device__ __forceinline__ v16bf frag_ptr(const bf16* rowbase, int k0, int kh) {
  const bf16* p = rowbase + k0 + kh;
  v8bf lo = *(const v8bf*)(p);
  v8bf hi = *(const v8bf*)(p + 16);
  return __builtin_shufflevector(lo, hi, 0, 1, 2, 3, 4, 5, 6, 7,
                                         8, 9, 10, 11, 12, 13, 14, 15);
}

__device__ __forceinline__ v16bf frag_rm(const bf16* base, int ld, int row0,
                                         int k0, int lane) {
  const bf16* p = base + (size_t)(row0 + (lane & 15)) * ld;
  return frag_ptr(p, k0, (lane >> 4) * 8);
}

__device__ __forceinline__ float sigf(float x) {
  return 1.0f / (1.0f + __expf(-x));
}

// ---------------------------------------------------------------------------
// 1. one-time conversions
// ---------------------------------------------------------------------------
__global__ void k_convert_emb(const float* __restrict__ emb,
                              bf16* __restrict__ embb) {
  size_t t = (size_t)blockIdx.x * blockDim.x + threadIdx.x;
  if (t >= (size_t)kVocab * kWDP) return;
  size_t r = t / kWDP;
  int d = (int)(t % kWDP);
  embb[t] = (bf16)((d < kWD) ? emb[r * kWD + d] : 0.0f);
}

__global__ void k_prep(const float* __restrict__ W_ih,
                       const float* __restrict__ W_hh,
                       bf16* __restrict__ wihb, bf16* __restrict__ whhb,
                       float* __restrict__ hc, float* __restrict__ cc,
                       bf16* __restrict__ hbf) {
  int t = blockIdx.x * 256 + threadIdx.x;
  const int N1 = kG * kWDP;  // 655360
  const int N2 = kG * kH;    // 1048576
  const int N3 = kR * kH;    // 262144
  if (t < N1) {
    int g = t / kWDP, d = t % kWDP;
    wihb[t] = (bf16)((d < kWD) ? W_ih[g * kWD + d] : 0.0f);
  } else if (t < N1 + N2) {
    int u = t - N1;
    whhb[u] = (bf16)W_hh[u];
  } else if (t < N1 + N2 + N3) {
    int u = t - N1 - N2;
    hc[u] = 0.0f;
    cc[u] = 0.0f;
    hbf[u] = (bf16)0.0f;
  }
}

// ---------------------------------------------------------------------------
// 2. x_proj = gather(emb) . W_ih^T + (b_ih + b_hh)      M=65536 N=2048 K=320
//    grid (4096 mtiles, 4) ; 8 waves/block, each wave: 1 A-tile x 4 N-tiles
//    row m = l*512 + R ; R = seq*256 + b ; token gathered per lane
// ---------------------------------------------------------------------------
__global__ void k_xproj_gemm(const int* __restrict__ question,
                             const int* __restrict__ sentence,
                             const bf16* __restrict__ embb,
                             const bf16* __restrict__ wihb,
                             const float* __restrict__ b_ih,
                             const float* __restrict__ b_hh,
                             bf16* __restrict__ xproj) {
  const int lane = threadIdx.x & 31;
  const int wave = threadIdx.x >> 5;
  const int m0 = blockIdx.x * 16;
  const int ntb = (blockIdx.y * 8 + wave) * 4;  // first of 4 n-tiles

  // per-lane gathered embedding row
  const int m = m0 + (lane & 15);
  const int l = m >> 9;      // timestep
  const int R = m & 511;     // fused row
  const int bb_ = R & 255;   // batch
  const int tok = (R < kB) ? question[bb_ * kSeq + l] : sentence[bb_ * kSeq + l];
  const bf16* arow = embb + (size_t)tok * kWDP;
  const int kh = (lane >> 4) * 8;

  constexpr int NK = kWDP / 32;  // 10
  v8f acc[4] = {};
  v16bf aR[2];
  v16bf bR[2][4];
  aR[0] = frag_ptr(arow, 0, kh);
#pragma unroll
  for (int j = 0; j < 4; ++j)
    bR[0][j] = frag_rm(wihb, kWDP, (ntb + j) * 16, 0, lane);

#pragma unroll
  for (int kc = 0; kc < NK; ++kc) {
    const int cur = kc & 1, nxt = cur ^ 1;
    if (kc + 1 < NK) {  // prefetch next chunk while WMMAs run on current
      const int k0 = (kc + 1) * 32;
      aR[nxt] = frag_ptr(arow, k0, kh);
#pragma unroll
      for (int j = 0; j < 4; ++j)
        bR[nxt][j] = frag_rm(wihb, kWDP, (ntb + j) * 16, k0, lane);
    }
#pragma unroll
    for (int j = 0; j < 4; ++j) acc[j] = wmma_bf16(aR[cur], bR[cur][j], acc[j]);
  }

  const int n_lane = lane & 15;
  const int mr_base = m0 + 8 * (lane >> 4);
#pragma unroll
  for (int j = 0; j < 4; ++j) {
    const int col = (ntb + j) * 16 + n_lane;
    const float bias = b_ih[col] + b_hh[col];
#pragma unroll
    for (int r = 0; r < 8; ++r) {
      xproj[(size_t)(mr_base + r) * kG + col] = (bf16)(acc[j][r] + bias);
    }
  }
}

// ---------------------------------------------------------------------------
// 3a. gates = h_bf . W_hh^T + x_proj[l]       M=512 N=2048 K=512
//     grid (32, 4) ; 8 waves/block x 4 n-tiles each ; pipelined K loop
// ---------------------------------------------------------------------------
__global__ void k_step_gemm(const bf16* __restrict__ hbf,
                            const bf16* __restrict__ whhb,
                            const bf16* __restrict__ xproj,
                            float* __restrict__ gates, int l) {
  const int lane = threadIdx.x & 31;
  const int wave = threadIdx.x >> 5;
  const int m0 = blockIdx.x * 16;
  const int ntb = (blockIdx.y * 8 + wave) * 4;

  constexpr int NK = kH / 32;  // 16
  v8f acc[4] = {};
  v16bf aR[2];
  v16bf bR[2][4];
  aR[0] = frag_rm(hbf, kH, m0, 0, lane);
#pragma unroll
  for (int j = 0; j < 4; ++j)
    bR[0][j] = frag_rm(whhb, kH, (ntb + j) * 16, 0, lane);

#pragma unroll
  for (int kc = 0; kc < NK; ++kc) {
    const int cur = kc & 1, nxt = cur ^ 1;
    if (kc + 1 < NK) {
      const int k0 = (kc + 1) * 32;
      aR[nxt] = frag_rm(hbf, kH, m0, k0, lane);
#pragma unroll
      for (int j = 0; j < 4; ++j)
        bR[nxt][j] = frag_rm(whhb, kH, (ntb + j) * 16, k0, lane);
    }
#pragma unroll
    for (int j = 0; j < 4; ++j) acc[j] = wmma_bf16(aR[cur], bR[cur][j], acc[j]);
  }

  const bf16* xp = xproj + (size_t)l * kR * kG;
  const int n_lane = lane & 15;
  const int mr_base = m0 + 8 * (lane >> 4);
#pragma unroll
  for (int j = 0; j < 4; ++j) {
    const int col = (ntb + j) * 16 + n_lane;
#pragma unroll
    for (int r = 0; r < 8; ++r) {
      const size_t idx = (size_t)(mr_base + r) * kG + col;
      gates[idx] = acc[j][r] + (float)xp[idx];
    }
  }
}

// ---------------------------------------------------------------------------
// 3b. LSTM cell update ; writes f32 h/c, bf16 h, hs[sb][l][h], hsT[sb][h][l]
// ---------------------------------------------------------------------------
__global__ void k_step_cell(const float* __restrict__ gates,
                            float* __restrict__ hc, float* __restrict__ cc,
                            bf16* __restrict__ hbf,
                            bf16* __restrict__ hsA, bf16* __restrict__ hsT,
                            int l) {
  int t = blockIdx.x * 256 + threadIdx.x;  // < 262144
  int r = t >> 9, h = t & 511;
  const float* g = gates + (size_t)r * kG;
  float ig = sigf(g[h]);
  float fg = sigf(g[kH + h]);
  float gg = tanhf(g[2 * kH + h]);
  float og = sigf(g[3 * kH + h]);
  float cn = fg * cc[t] + ig * gg;
  float hn = og * tanhf(cn);
  cc[t] = cn;
  hc[t] = hn;
  hbf[t] = (bf16)hn;
  // r already equals seq*256 + batch
  hsA[((size_t)r * kSeq + l) * kH + h] = (bf16)hn;
  hsT[((size_t)r * kH + h) * kSeq + l] = (bf16)hn;
}

// ---------------------------------------------------------------------------
// 4a. attn[b][q][s] = sum_h _i[q][h] * _t[s][h]   (K=512, WMMA, pipelined)
//     grid (256 batch, 8 qtile) ; wave = stile
// ---------------------------------------------------------------------------
__global__ void k_attn(const bf16* __restrict__ hsA, float* __restrict__ attn) {
  const int lane = threadIdx.x & 31;
  const int wave = threadIdx.x >> 5;
  const int b = blockIdx.x;
  const bf16* Aq = hsA + (size_t)b * kSeq * kH;          // question (_i)
  const bf16* Bt = hsA + ((size_t)kB + b) * kSeq * kH;   // sentence (_t) = B^T
  const int q0 = blockIdx.y * 16, s0 = wave * 16;

  constexpr int NK = kH / 32;  // 16
  v8f acc = {};
  v16bf aR[2], bR[2];
  aR[0] = frag_rm(Aq, kH, q0, 0, lane);
  bR[0] = frag_rm(Bt, kH, s0, 0, lane);
#pragma unroll
  for (int kc = 0; kc < NK; ++kc) {
    const int cur = kc & 1, nxt = cur ^ 1;
    if (kc + 1 < NK) {
      const int k0 = (kc + 1) * 32;
      aR[nxt] = frag_rm(Aq, kH, q0, k0, lane);
      bR[nxt] = frag_rm(Bt, kH, s0, k0, lane);
    }
    acc = wmma_bf16(aR[cur], bR[cur], acc);
  }
  float* dst = attn + (size_t)b * kSeq * kSeq;
  const int n = lane & 15, mrb = q0 + 8 * (lane >> 4);
#pragma unroll
  for (int r = 0; r < 8; ++r) dst[(size_t)(mrb + r) * kSeq + s0 + n] = acc[r];
}

// 4b. row softmax over s -> P1 bf16 ; one wave per row
__global__ void k_softmax_row(const float* __restrict__ attn,
                              bf16* __restrict__ P1) {
  const int lane = threadIdx.x & 31;
  const int row = blockIdx.x * 8 + (threadIdx.x >> 5);  // b*128 + q
  const float* a = attn + (size_t)row * kSeq;
  float x0 = a[lane], x1 = a[lane + 32], x2 = a[lane + 64], x3 = a[lane + 96];
  float m = fmaxf(fmaxf(x0, x1), fmaxf(x2, x3));
  for (int off = 16; off > 0; off >>= 1) m = fmaxf(m, __shfl_xor(m, off));
  float e0 = __expf(x0 - m), e1 = __expf(x1 - m);
  float e2 = __expf(x2 - m), e3 = __expf(x3 - m);
  float s = e0 + e1 + e2 + e3;
  for (int off = 16; off > 0; off >>= 1) s += __shfl_xor(s, off);
  float inv = 1.0f / s;
  bf16* p = P1 + (size_t)row * kSeq;
  p[lane] = (bf16)(e0 * inv);
  p[lane + 32] = (bf16)(e1 * inv);
  p[lane + 64] = (bf16)(e2 * inv);
  p[lane + 96] = (bf16)(e3 * inv);
}

// 4c. column softmax over q, stored transposed: P2[b][s][q]
__global__ void k_softmax_col(const float* __restrict__ attn,
                              bf16* __restrict__ P2) {
  const int lane = threadIdx.x & 31;
  const int col = blockIdx.x * 8 + (threadIdx.x >> 5);  // b*128 + s
  const int b = col >> 7, s = col & 127;
  const float* a = attn + (size_t)b * kSeq * kSeq + s;
  float x0 = a[(size_t)lane * kSeq];
  float x1 = a[(size_t)(lane + 32) * kSeq];
  float x2 = a[(size_t)(lane + 64) * kSeq];
  float x3 = a[(size_t)(lane + 96) * kSeq];
  float m = fmaxf(fmaxf(x0, x1), fmaxf(x2, x3));
  for (int off = 16; off > 0; off >>= 1) m = fmaxf(m, __shfl_xor(m, off));
  float e0 = __expf(x0 - m), e1 = __expf(x1 - m);
  float e2 = __expf(x2 - m), e3 = __expf(x3 - m);
  float s4 = e0 + e1 + e2 + e3;
  for (int off = 16; off > 0; off >>= 1) s4 += __shfl_xor(s4, off);
  float inv = 1.0f / s4;
  bf16* p = P2 + ((size_t)b * kSeq + s) * kSeq;
  p[lane] = (bf16)(e0 * inv);
  p[lane + 32] = (bf16)(e1 * inv);
  p[lane + 64] = (bf16)(e2 * inv);
  p[lane + 96] = (bf16)(e3 * inv);
}

// 4d. fused  out = P(128x128) . V(128x512)  +  max over the 128 rows.
//     grid (256 batch, 32 htile) ; 8 waves cover all 128 rows -> direct max.
//     Bt = transposed hidden states [H][L] (rows = output columns), K = 128.
__global__ void k_out_max(const bf16* __restrict__ P,
                          const bf16* __restrict__ hsTsel,
                          float* __restrict__ outmax) {
  __shared__ float red[8][16];
  const int lane = threadIdx.x & 31;
  const int wave = threadIdx.x >> 5;
  const int b = blockIdx.x, ht = blockIdx.y;
  const bf16* A = P + (size_t)b * kSeq * kSeq;
  const bf16* Bt = hsTsel + (size_t)b * kH * kSeq;
  const int m0 = wave * 16, n0 = ht * 16;

  constexpr int NK = kSeq / 32;  // 4
  v8f acc = {};
  v16bf aR[2], bR[2];
  aR[0] = frag_rm(A, kSeq, m0, 0, lane);
  bR[0] = frag_rm(Bt, kSeq, n0, 0, lane);
#pragma unroll
  for (int kc = 0; kc < NK; ++kc) {
    const int cur = kc & 1, nxt = cur ^ 1;
    if (kc + 1 < NK) {
      const int k0 = (kc + 1) * 32;
      aR[nxt] = frag_rm(A, kSeq, m0, k0, lane);
      bR[nxt] = frag_rm(Bt, kSeq, n0, k0, lane);
    }
    acc = wmma_bf16(aR[cur], bR[cur], acc);
  }
  float m = acc[0];
#pragma unroll
  for (int r = 1; r < 8; ++r) m = fmaxf(m, acc[r]);
  m = fmaxf(m, __shfl_xor(m, 16));  // combine rows m..m+7 with m+8..m+15
  if (lane < 16) red[wave][lane] = m;
  __syncthreads();
  if (threadIdx.x < 16) {
    float v = red[0][threadIdx.x];
#pragma unroll
    for (int w = 1; w < 8; ++w) v = fmaxf(v, red[w][threadIdx.x]);
    outmax[(size_t)b * kH + n0 + threadIdx.x] = v;
  }
}

// ---------------------------------------------------------------------------
// 5. classifier: out[b][j] = sigmoid(b_last[j] + feat . W_last[j])
//    feat = [hq, hs, hq-hs, hq*hs]  (2048)
// ---------------------------------------------------------------------------
__global__ void k_final(const float* __restrict__ hq,
                        const float* __restrict__ hsm,
                        const float* __restrict__ W_last,
                        const float* __restrict__ b_last,
                        float* __restrict__ out) {
  __shared__ float s0[256];
  __shared__ float s1[256];
  const int b = blockIdx.x, t = threadIdx.x;
  float p0 = 0.0f, p1 = 0.0f;
  for (int k = t; k < 4 * kH; k += 256) {
    int seg = k >> 9, h = k & 511;
    float a = hq[(size_t)b * kH + h];
    float c = hsm[(size_t)b * kH + h];
    float f = (seg == 0) ? a : (seg == 1) ? c : (seg == 2) ? (a - c) : (a * c);
    p0 += f * W_last[k];
    p1 += f * W_last[4 * kH + k];
  }
  s0[t] = p0;
  s1[t] = p1;
  __syncthreads();
  for (int off = 128; off > 0; off >>= 1) {
    if (t < off) {
      s0[t] += s0[t + off];
      s1[t] += s1[t + off];
    }
    __syncthreads();
  }
  if (t == 0) {
    out[b * 2 + 0] = sigf(s0[0] + b_last[0]);
    out[b * 2 + 1] = sigf(s1[0] + b_last[1]);
  }
}

// ---------------------------------------------------------------------------
// host launcher
// ---------------------------------------------------------------------------
extern "C" void kernel_launch(void* const* d_in, const int* in_sizes, int n_in,
                              void* d_out, int out_size, void* d_ws,
                              size_t ws_size, hipStream_t stream) {
  (void)in_sizes; (void)n_in; (void)out_size; (void)ws_size;
  const int*   question = (const int*)d_in[0];
  const int*   sentence = (const int*)d_in[1];
  const float* emb    = (const float*)d_in[2];
  const float* W_ih   = (const float*)d_in[3];
  const float* b_ih   = (const float*)d_in[4];
  const float* W_hh   = (const float*)d_in[5];
  const float* b_hh   = (const float*)d_in[6];
  const float* W_last = (const float*)d_in[7];
  const float* b_last = (const float*)d_in[8];
  float* out = (float*)d_out;

  char* wsb = (char*)d_ws;
  size_t off = 0;
  auto alloc = [&](size_t bytes) -> void* {
    void* p = wsb + off;
    off = (off + bytes + 255) & ~(size_t)255;
    return p;
  };
  bf16*  embb  = (bf16*) alloc((size_t)kVocab * kWDP * 2);   // 58.6 MB
  bf16*  wihb  = (bf16*) alloc((size_t)kG * kWDP * 2);       // 1.3 MB
  bf16*  whhb  = (bf16*) alloc((size_t)kG * kH * 2);         // 2.1 MB
  bf16*  xproj = (bf16*) alloc((size_t)kMXP * kG * 2);       // 268 MB
  float* gates = (float*)alloc((size_t)kR * kG * 4);         // 4 MB
  float* hc    = (float*)alloc((size_t)kR * kH * 4);
  float* cc    = (float*)alloc((size_t)kR * kH * 4);
  bf16*  hbf   = (bf16*) alloc((size_t)kR * kH * 2);
  bf16*  hsA   = (bf16*) alloc((size_t)2 * kB * kSeq * kH * 2);  // 67 MB
  bf16*  hsT   = (bf16*) alloc((size_t)2 * kB * kH * kSeq * 2);  // 67 MB
  float* attn  = (float*)alloc((size_t)kB * kSeq * kSeq * 4);    // 16.8 MB
  bf16*  P1    = (bf16*) alloc((size_t)kB * kSeq * kSeq * 2);
  bf16*  P2    = (bf16*) alloc((size_t)kB * kSeq * kSeq * 2);
  float* hq    = (float*)alloc((size_t)kB * kH * 4);
  float* hsm   = (float*)alloc((size_t)kB * kH * 4);

  // 1. conversions + zero-init state
  {
    size_t n = (size_t)kVocab * kWDP;
    k_convert_emb<<<dim3((unsigned)((n + 255) / 256)), 256, 0, stream>>>(emb, embb);
  }
  {
    int n = kG * kWDP + kG * kH + kR * kH;  // 1,966,080
    k_prep<<<dim3((unsigned)((n + 255) / 256)), 256, 0, stream>>>(
        W_ih, W_hh, wihb, whhb, hc, cc, hbf);
  }

  // 2. input projection GEMM (gathered embeddings)
  k_xproj_gemm<<<dim3(kMXP / 16, 4), 256, 0, stream>>>(
      question, sentence, embb, wihb, b_ih, b_hh, xproj);

  // 3. LSTM recurrence (question + sentence fused: 512 rows per step)
  for (int l = 0; l < kSeq; ++l) {
    k_step_gemm<<<dim3(kR / 16, 4), 256, 0, stream>>>(hbf, whhb, xproj, gates, l);
    k_step_cell<<<dim3((kR * kH) / 256), 256, 0, stream>>>(
        gates, hc, cc, hbf, hsA, hsT, l);
  }

  // 4. attention
  k_attn<<<dim3(kB, kSeq / 16), 256, 0, stream>>>(hsA, attn);
  k_softmax_row<<<dim3((kB * kSeq) / 8), 256, 0, stream>>>(attn, P1);
  k_softmax_col<<<dim3((kB * kSeq) / 8), 256, 0, stream>>>(attn, P2);
  // out1 = P1 . _t  -> max over q  (V^T = sentence half of hsT)
  k_out_max<<<dim3(kB, kH / 16), 256, 0, stream>>>(
      P1, hsT + (size_t)kB * kH * kSeq, hq);
  // out2 = P2 . _i  -> max over s  (V^T = question half of hsT)
  k_out_max<<<dim3(kB, kH / 16), 256, 0, stream>>>(P2, hsT, hsm);

  // 5. classifier
  k_final<<<dim3(kB), 256, 0, stream>>>(hq, hsm, W_last, b_last, out);
}
